// DynamicGraphAttentionLayer_67422396613191
// MI455X (gfx1250) — matrix operand order repaired and checked by reading, not compiled, not run on hardware
//
#include <hip/hip_runtime.h>
#include <hip/hip_bf16.h>
#include <stdint.h>

// ---------------------------------------------------------------------------
// GATConv for MI455X (gfx1250, wave32).
// Phases:
//   K_init  : out = bias broadcast, mkey = 0, s = 0
//   K_cvtx  : x (f32) -> x_f16 row-major
//   K_swzw  : W (f32) -> f16, pre-swizzled into WMMA B-fragment layout
//   K_gemm  : h = x @ W via v_wmma_f32_16x16x32_f16 (f32 accumulate)
//   K_att   : a_s/a_d per (node, head), wave-per-dot with shfl reduction
//   K_elog  : per-edge leakyrelu logits + segment-max via u32-ordered atomicMax
//   K_eexp  : per-edge exp(e - m) + segment-sum via atomicAdd
//   K_aggr  : out[dst] += alpha * h[src], 64 lanes/edge, float4 + atomicAdd f32
// h (51 MB) and out (51 MB) both fit in the 192 MB L2 -> the scatter phase is
// L2-bandwidth bound, so plain f32 atomics are the right tool.
// ---------------------------------------------------------------------------

#define NN    50000      // nodes
#define NE    1600000    // edges (before self loops)
#define ETOT  (NE + NN)  // edges + self loops
#define IND   128        // in dim
#define HF    256        // heads * out dim
#define FD    64         // out dim per head
#define NH    4          // heads

typedef _Float16 v8h  __attribute__((ext_vector_type(8)));
typedef _Float16 v16h __attribute__((ext_vector_type(16)));
typedef float    v8f  __attribute__((ext_vector_type(8)));

// ---------------- init: out = bias, mkey = 0, s = 0 ------------------------
__global__ __launch_bounds__(256) void K_init(float* __restrict__ out,
                                              const float* __restrict__ bias,
                                              unsigned int* __restrict__ mkey,
                                              float* __restrict__ s) {
  int i = blockIdx.x * 256 + threadIdx.x;
  if (i < NN * HF) out[i] = bias[i & (HF - 1)];
  if (i < NN * NH) { mkey[i] = 0u; s[i] = 0.0f; }
}

// ---------------- x -> f16 -------------------------------------------------
__global__ __launch_bounds__(256) void K_cvtx(const float* __restrict__ x,
                                              _Float16* __restrict__ xh) {
  int i = blockIdx.x * 256 + threadIdx.x;
  if (i < NN * IND) xh[i] = (_Float16)x[i];
}

// ---------------- W -> f16, swizzled to B-fragment layout ------------------
// layout: [kt(4)][nt(16)][lane(32)][j(16)] f16; element j of lane l in tile
// (kt, nt) is B[k][n] with k = kt*32 + (l/16)*16 + j, n = nt*16 + (l%16).
__global__ __launch_bounds__(256) void K_swzw(const float* __restrict__ W,
                                              _Float16* __restrict__ wsw) {
  int idx = blockIdx.x * 256 + threadIdx.x;
  if (idx >= 4 * 16 * 32 * 16) return;
  int j    = idx & 15;
  int lane = (idx >> 4) & 31;
  int nt   = (idx >> 9) & 15;
  int kt   = idx >> 13;
  int k = kt * 32 + (lane >> 4) * 16 + j;
  int n = nt * 16 + (lane & 15);
  wsw[idx] = (_Float16)W[k * HF + n];
}

// ---------------- GEMM: h[50000x256] = x_f16 @ W_f16 (f32 acc) -------------
__global__ __launch_bounds__(256) void K_gemm(const _Float16* __restrict__ xh,
                                              const _Float16* __restrict__ wsw,
                                              float* __restrict__ h) {
  const int lane  = threadIdx.x & 31;
  const int wave  = threadIdx.x >> 5;          // 0..7
  const int mbase = blockIdx.x * 16;           // 3125 blocks * 16 rows = 50000
  const int rowA  = mbase + (lane & 15);
  const int kseg  = (lane >> 4) * 16;          // A-fragment K sub-offset
  const int nt0 = wave, nt1 = wave + 8;        // two 16-wide column tiles

  v8f c0 = {}; v8f c1 = {};
  #pragma unroll
  for (int kt = 0; kt < 4; ++kt) {
    // A fragment: two contiguous 16B chunks per lane (row-major x_f16)
    const _Float16* ap = xh + rowA * IND + kt * 32 + kseg;
    v8h alo = *(const v8h*)ap;
    v8h ahi = *(const v8h*)(ap + 8);
    v16h a = __builtin_shufflevector(alo, ahi, 0,1,2,3,4,5,6,7,8,9,10,11,12,13,14,15);
    // B fragments: one contiguous 32B chunk per lane from swizzled W
    v16h b0 = *(const v16h*)(wsw + (((kt * 16 + nt0) * 32 + lane) << 4));
    v16h b1 = *(const v16h*)(wsw + (((kt * 16 + nt1) * 32 + lane) << 4));
    c0 = __builtin_amdgcn_wmma_f32_16x16x32_f16(false, a, false, b0, (short)0, c0, false, false);
    c1 = __builtin_amdgcn_wmma_f32_16x16x32_f16(false, a, false, b1, (short)0, c1, false, false);
  }
  // D layout: lane l, vgpr v -> row = v + (l/16)*8, col = l%16
  const int rbase = mbase + (lane >> 4) * 8;
  const int col0  = nt0 * 16 + (lane & 15);
  const int col1  = nt1 * 16 + (lane & 15);
  #pragma unroll
  for (int v = 0; v < 8; ++v) {
    h[(rbase + v) * HF + col0] = c0[v];
    h[(rbase + v) * HF + col1] = c1[v];
  }
}

// ---------------- per-(node,head) attention logits -------------------------
__global__ __launch_bounds__(256) void K_att(const float* __restrict__ h,
                                             const float* __restrict__ att_src,
                                             const float* __restrict__ att_dst,
                                             float* __restrict__ a_s,
                                             float* __restrict__ a_d) {
  int wave = (blockIdx.x * 256 + threadIdx.x) >> 5;   // one wave per (node,head)
  int lane = threadIdx.x & 31;
  if (wave >= NN * NH) return;
  int node = wave >> 2, hd = wave & 3;
  const float* hp = h + node * HF + hd * FD;
  const float* as = att_src + hd * FD;
  const float* ad = att_dst + hd * FD;
  float h0 = hp[lane], h1 = hp[lane + 32];
  float ss = h0 * as[lane] + h1 * as[lane + 32];
  float sd = h0 * ad[lane] + h1 * ad[lane + 32];
  #pragma unroll
  for (int off = 16; off > 0; off >>= 1) {
    ss += __shfl_xor(ss, off, 32);
    sd += __shfl_xor(sd, off, 32);
  }
  if (lane == 0) { a_s[wave] = ss; a_d[wave] = sd; }
}

// monotone float->u32 key (order-preserving), so u32 atomicMax == float max
__device__ __forceinline__ unsigned int f2key(float v) {
  unsigned int u = __float_as_uint(v);
  return (u & 0x80000000u) ? ~u : (u | 0x80000000u);
}
__device__ __forceinline__ float key2f(unsigned int k) {
  return (k & 0x80000000u) ? __uint_as_float(k & 0x7fffffffu) : __uint_as_float(~k);
}

__device__ __forceinline__ void edge_src_dst(const long long* ei, int e, int& src, int& dst) {
  if (e < NE) { src = (int)ei[e]; dst = (int)ei[NE + e]; }
  else        { src = dst = e - NE; }          // appended self loops
}

// ---------------- per-edge logits + segment max ----------------------------
__global__ __launch_bounds__(256) void K_elog(const long long* __restrict__ ei,
                                              const float* __restrict__ a_s,
                                              const float* __restrict__ a_d,
                                              float* __restrict__ ebuf,
                                              unsigned int* __restrict__ mkey) {
  int e = blockIdx.x * 256 + threadIdx.x;
  if (e >= ETOT) return;
  int src, dst; edge_src_dst(ei, e, src, dst);
  #pragma unroll
  for (int hd = 0; hd < NH; ++hd) {
    float v = a_s[src * NH + hd] + a_d[dst * NH + hd];
    v = v > 0.0f ? v : 0.2f * v;               // leaky relu
    ebuf[e * NH + hd] = v;
    atomicMax(&mkey[dst * NH + hd], f2key(v));
  }
}

// ---------------- per-edge exp + segment sum -------------------------------
__global__ __launch_bounds__(256) void K_eexp(const long long* __restrict__ ei,
                                              float* __restrict__ ebuf,
                                              const unsigned int* __restrict__ mkey,
                                              float* __restrict__ s) {
  int e = blockIdx.x * 256 + threadIdx.x;
  if (e >= ETOT) return;
  int src, dst; edge_src_dst(ei, e, src, dst);
  (void)src;
  #pragma unroll
  for (int hd = 0; hd < NH; ++hd) {
    float m  = key2f(mkey[dst * NH + hd]);
    float ex = __expf(ebuf[e * NH + hd] - m);
    ebuf[e * NH + hd] = ex;
    atomicAdd(&s[dst * NH + hd], ex);
  }
}

// ---------------- weighted scatter-aggregate -------------------------------
// 64 lanes per edge, 4 features (one float4) per lane; h and out live in L2.
__global__ __launch_bounds__(256) void K_aggr(const long long* __restrict__ ei,
                                              const float* __restrict__ h,
                                              const float* __restrict__ ebuf,
                                              const float* __restrict__ s,
                                              float* __restrict__ out) {
  long long gid = (long long)blockIdx.x * 256 + threadIdx.x;
  int e = (int)(gid >> 6);
  int t = (int)(gid & 63);
  if (e >= ETOT) return;
  int src, dst; edge_src_dst(ei, e, src, dst);
  int f  = t << 2;                              // feature base 0..252
  int hd = t >> 4;                              // 16 lanes per head (4*16=64 feats)
  float alpha = ebuf[e * NH + hd] / (s[dst * NH + hd] + 1e-16f);
  const float4 hv = *(const float4*)(h + src * HF + f);
  float* op = out + dst * HF + f;
  atomicAdd(op + 0, hv.x * alpha);
  atomicAdd(op + 1, hv.y * alpha);
  atomicAdd(op + 2, hv.z * alpha);
  atomicAdd(op + 3, hv.w * alpha);
}

// ---------------------------------------------------------------------------
extern "C" void kernel_launch(void* const* d_in, const int* in_sizes, int n_in,
                              void* d_out, int out_size, void* d_ws, size_t ws_size,
                              hipStream_t stream) {
  const float*     x    = (const float*)d_in[0];
  const long long* ei   = (const long long*)d_in[1];   // int64 [2, NE]
  const float*     W    = (const float*)d_in[2];
  const float*     atts = (const float*)d_in[3];
  const float*     attd = (const float*)d_in[4];
  const float*     bias = (const float*)d_in[5];
  float* out = (float*)d_out;

  // workspace layout (bytes, all 256B aligned)
  char* ws = (char*)d_ws;
  _Float16*     xh   = (_Float16*)(ws);                          // 12,800,000
  _Float16*     wsw  = (_Float16*)(ws + 12800000);               //     65,536
  float*        h    = (float*)   (ws + 12865536);               // 51,200,000
  float*        a_s  = (float*)   (ws + 64065536);               //    800,000
  float*        a_d  = (float*)   (ws + 64865536);               //    800,000
  unsigned int* mkey = (unsigned int*)(ws + 65665536);           //    800,000
  float*        ssum = (float*)   (ws + 66465536);               //    800,000
  float*        ebuf = (float*)   (ws + 67265536);               // 26,400,000
  (void)in_sizes; (void)n_in; (void)out_size; (void)ws_size;

  K_init<<<(NN * HF + 255) / 256, 256, 0, stream>>>(out, bias, mkey, ssum);
  K_cvtx<<<(NN * IND + 255) / 256, 256, 0, stream>>>(x, xh);
  K_swzw<<<(4 * 16 * 32 * 16 + 255) / 256, 256, 0, stream>>>(W, wsw);
  K_gemm<<<NN / 16, 256, 0, stream>>>(xh, wsw, h);
  K_att <<<(NN * NH * 32 + 255) / 256, 256, 0, stream>>>(h, atts, attd, a_s, a_d);
  K_elog<<<(ETOT + 255) / 256, 256, 0, stream>>>(ei, a_s, a_d, ebuf, mkey);
  K_eexp<<<(ETOT + 255) / 256, 256, 0, stream>>>(ei, ebuf, mkey, ssum);
  long long aggThreads = (long long)ETOT * 64;
  K_aggr<<<(unsigned int)((aggThreads + 255) / 256), 256, 0, stream>>>(ei, h, ebuf, ssum, out);
}
